// SeqCopyGenerator_69801808495265
// MI455X (gfx1250) — matrix operand order repaired and checked by reading, not compiled, not run on hardware
//
#include <hip/hip_runtime.h>
#include <hip/hip_bf16.h>
#include <math.h>

// ---------------------------------------------------------------------------
// SeqCopyGenerator for MI455X (gfx1250, wave32, WMMA + async-to-LDS).
//   cvt:   ctx, word_emb, word_emb_tgt -> bf16 (A/B operands for WMMA)
//   G1-3:  Ycpy/Lspt/Lnrm = ctx_bf @ W_* + b_*       (bf16 WMMA, f32 B staged+cvt)
//   rowdot: pcpy[t] = <Ycpy[t,:], ctx[t,:]>
//   rowproc_*: log(softmax+eps)+gumbel -> softmax -> argmax, register-resident;
//              sft rows written as bf16 (half the HBM traffic of f32)
//   G4/G5: sft_bf @ word_emb(_tgt)_bf               (bf16 WMMA, split-K + f32 atomics)
//   finalize: per-token select + embedding blend
// GEMM double-buffers LDS: tile i+1 is staged (A via GLOBAL_LOAD_ASYNC_TO_LDS_B128,
// B via vector loads + transposed ds stores) while tile i feeds the WMMAs.
// ---------------------------------------------------------------------------

#define NT   1024      // B*L tokens
#define HS   1024
#define NWRD 32000
#define NTGT 2000
#define DIM  128
#define EPSV 1e-12f

typedef __bf16 bf16_t;
typedef bf16_t v16bf __attribute__((ext_vector_type(16)));
typedef bf16_t v8bf  __attribute__((ext_vector_type(8)));
typedef float  v8f   __attribute__((ext_vector_type(8)));

union ABfrag { v16bf v; v8bf h[2]; };

__device__ __forceinline__ unsigned int f2bf(float f) {
  unsigned int u = __float_as_uint(f);
  return (u + 0x7FFFu + ((u >> 16) & 1u)) >> 16;   // round-nearest-even, low 16 bits
}

// ---------------------------------------------------------------------------
// GEMM: C[M,N] (+)= A_bf16[M,K] * B[K,N] (+ bias). A is bf16; B is f32 or bf16.
// Block tile 128x128x32, 8 waves (2M x 4N), 4x2 v_wmma_f32_16x16x32_bf16 per wave.
// Double-buffered LDS; A staged by the async-to-LDS engine (ASYNCcnt).
// gridDim.z>1 => split-K partial sums via global_atomic_add_f32 (C pre-zeroed).
// ---------------------------------------------------------------------------
#define BM 128
#define BN 128
#define BKS 32
#define LDSS 56    // LDS row stride in bf16: 112B, 16B-aligned, conflict-free

template <bool B_IS_BF16>
__global__ __launch_bounds__(256, 2)
void gemm_wmma(const unsigned short* __restrict__ A, const void* __restrict__ Bv,
               const float* __restrict__ bias, float* __restrict__ C,
               int M, int N, int K, int lda, int ldb, int ldc, int Kc)
{
  __shared__ unsigned short As[2][BM * LDSS];   // A tiles, row-major [128][32]
  __shared__ unsigned short Bs[2][BN * LDSS];   // B tiles, TRANSPOSED [n][k]

  const int tid = threadIdx.x;
  const int n0 = blockIdx.x * BN;
  const int m0 = blockIdx.y * BM;
  const int kstart = blockIdx.z * Kc;
  const int kend = min(K, kstart + Kc);
  const bool split = (gridDim.z > 1);
  const bool nfull = (n0 + BN <= N);

  const int lane = tid & 31;
  const int l16  = lane & 15;
  const int hi   = lane >> 4;
  const int wid  = tid >> 5;
  const int wm   = wid & 1;     // 64-row half
  const int wn   = wid >> 1;    // 32-col quarter

  v8f acc[4][2];
  const v8f vzero = {0.f,0.f,0.f,0.f,0.f,0.f,0.f,0.f};
  #pragma unroll
  for (int i = 0; i < 4; ++i)
    #pragma unroll
    for (int j = 0; j < 2; ++j) acc[i][j] = vzero;

  // ---- stage one K-tile into LDS buffer `buf` ----
  auto stage = [&](int buf, int kb) {
    const bool kfull = (kb + BKS <= kend);
    // A tile 128x32 bf16
    if (kfull) {
      // async DMA: 2 x 16B segments per thread (512 x b128 = 8KB tile), ASYNCcnt
      #pragma unroll
      for (int i = 0; i < 2; ++i) {
        const int e   = tid + (i << 8);
        const int row = e >> 2;
        const int seg = e & 3;
        const unsigned short* gp =
            A + (size_t)(m0 + row) * (size_t)lda + kb + (seg << 3);
        const unsigned int laddr =
            (unsigned int)(unsigned long long)&As[buf][row * LDSS + (seg << 3)];
        asm volatile("global_load_async_to_lds_b128 %0, %1, off"
                     :: "v"(laddr), "v"((unsigned long long)gp) : "memory");
      }
    } else {
      #pragma unroll
      for (int i = 0; i < 2; ++i) {
        const int e   = tid + (i << 8);
        const int row = e >> 2;
        const int seg = e & 3;
        #pragma unroll
        for (int j = 0; j < 8; ++j) {
          const int k = kb + (seg << 3) + j;
          As[buf][row * LDSS + (seg << 3) + j] =
              (k < kend) ? A[(size_t)(m0 + row) * (size_t)lda + k]
                         : (unsigned short)0;
        }
      }
    }
    // B tile 32x128 -> transposed Bs[n][k]
    if (B_IS_BF16) {
      const unsigned short* B = (const unsigned short*)Bv;
      if (kfull & nfull) {
        const int kp = tid >> 4;            // 0..15 K-pairs
        const int n8 = (tid & 15) << 3;     // 0..120
        const unsigned short* bp0 =
            B + (size_t)(kb + (kp << 1)) * (size_t)ldb + n0 + n8;
        const uint4 r0 = *(const uint4*)bp0;
        const uint4 r1 = *(const uint4*)(bp0 + ldb);
        unsigned short* d = &Bs[buf][n8 * LDSS + (kp << 1)];
        *(unsigned int*)(d + 0 * LDSS) = (r0.x & 0xFFFFu) | (r1.x << 16);
        *(unsigned int*)(d + 1 * LDSS) = (r0.x >> 16)     | (r1.x & 0xFFFF0000u);
        *(unsigned int*)(d + 2 * LDSS) = (r0.y & 0xFFFFu) | (r1.y << 16);
        *(unsigned int*)(d + 3 * LDSS) = (r0.y >> 16)     | (r1.y & 0xFFFF0000u);
        *(unsigned int*)(d + 4 * LDSS) = (r0.z & 0xFFFFu) | (r1.z << 16);
        *(unsigned int*)(d + 5 * LDSS) = (r0.z >> 16)     | (r1.z & 0xFFFF0000u);
        *(unsigned int*)(d + 6 * LDSS) = (r0.w & 0xFFFFu) | (r1.w << 16);
        *(unsigned int*)(d + 7 * LDSS) = (r0.w >> 16)     | (r1.w & 0xFFFF0000u);
      } else {
        const int n   = tid & 127;
        const int col = n0 + n;
        #pragma unroll
        for (int j = 0; j < 8; ++j) {
          const int kp = ((tid >> 7) << 3) + j;
          const int k  = kb + (kp << 1);
          unsigned int b0 = 0, b1 = 0;
          if (col < N) {
            if (k     < kend) b0 = B[(size_t)k       * (size_t)ldb + col];
            if (k + 1 < kend) b1 = B[(size_t)(k + 1) * (size_t)ldb + col];
          }
          *(unsigned int*)(&Bs[buf][n * LDSS + (kp << 1)]) = b0 | (b1 << 16);
        }
      }
    } else {
      const float* B = (const float*)Bv;
      if (kfull & nfull) {
        #pragma unroll
        for (int i = 0; i < 2; ++i) {
          const int u  = tid + (i << 8);
          const int kp = u >> 5;             // 0..15
          const int c4 = (u & 31) << 2;      // 0..124
          const float* bp0 =
              B + (size_t)(kb + (kp << 1)) * (size_t)ldb + n0 + c4;
          const float4 r0 = *(const float4*)bp0;
          const float4 r1 = *(const float4*)(bp0 + ldb);
          unsigned short* d = &Bs[buf][c4 * LDSS + (kp << 1)];
          *(unsigned int*)(d + 0 * LDSS) = f2bf(r0.x) | (f2bf(r1.x) << 16);
          *(unsigned int*)(d + 1 * LDSS) = f2bf(r0.y) | (f2bf(r1.y) << 16);
          *(unsigned int*)(d + 2 * LDSS) = f2bf(r0.z) | (f2bf(r1.z) << 16);
          *(unsigned int*)(d + 3 * LDSS) = f2bf(r0.w) | (f2bf(r1.w) << 16);
        }
        if (kb + BKS < kend)                                // global_prefetch_b8
          __builtin_prefetch(&B[(size_t)(kb + BKS) * (size_t)ldb + n0 + (tid >> 1)], 0, 0);
      } else {
        const int n   = tid & 127;
        const int col = n0 + n;
        #pragma unroll
        for (int j = 0; j < 8; ++j) {
          const int kp = ((tid >> 7) << 3) + j;
          const int k  = kb + (kp << 1);
          float b0 = 0.f, b1 = 0.f;
          if (col < N) {
            if (k     < kend) b0 = B[(size_t)k       * (size_t)ldb + col];
            if (k + 1 < kend) b1 = B[(size_t)(k + 1) * (size_t)ldb + col];
          }
          *(unsigned int*)(&Bs[buf][n * LDSS + (kp << 1)]) =
              f2bf(b0) | (f2bf(b1) << 16);
        }
      }
    }
  };

  // ---- consume one staged tile: 8 WMMAs ----
  // A 16x32 bf16: lanes 0-15 row=l16 hold K{0..7,16..23}; lanes 16-31 K{8..15,24..31}
  // B 32x16 bf16 (transposed LDS): lanes 0-15 col=l16 hold K{0..7,16..23}; lanes 16-31 K{8..15,24..31}
  auto compute = [&](int buf) {
    ABfrag afr[4], bfr[2];
    #pragma unroll
    for (int mt = 0; mt < 4; ++mt) {
      const int row = wm * 64 + mt * 16 + l16;
      const v8bf* p = (const v8bf*)&As[buf][row * LDSS];
      afr[mt].h[0] = p[hi];
      afr[mt].h[1] = p[hi + 2];
    }
    #pragma unroll
    for (int nt = 0; nt < 2; ++nt) {
      const int row = wn * 32 + nt * 16 + l16;
      const v8bf* q = (const v8bf*)&Bs[buf][row * LDSS];
      bfr[nt].h[0] = q[hi];
      bfr[nt].h[1] = q[hi + 2];
    }
    #pragma unroll
    for (int mt = 0; mt < 4; ++mt)
      #pragma unroll
      for (int nt = 0; nt < 2; ++nt)
        acc[mt][nt] = __builtin_amdgcn_wmma_f32_16x16x32_bf16(
            false, afr[mt].v, false, bfr[nt].v, (short)0, acc[mt][nt], false, false);
  };

  const int nIter = (kend > kstart) ? (kend - kstart + BKS - 1) / BKS : 0;
  if (nIter > 0) {
    stage(0, kstart);
    asm volatile("s_wait_asynccnt 0x0" ::: "memory");
    __syncthreads();
    int it = 0;
    for (;;) {
      // parity 0: compute buf 0, stage buf 1
      if (it + 1 < nIter) stage(1, kstart + (it + 1) * BKS);
      compute(0);
      asm volatile("s_wait_asynccnt 0x0" ::: "memory");
      __syncthreads();
      if (++it == nIter) break;
      // parity 1: compute buf 1, stage buf 0
      if (it + 1 < nIter) stage(0, kstart + (it + 1) * BKS);
      compute(1);
      asm volatile("s_wait_asynccnt 0x0" ::: "memory");
      __syncthreads();
      if (++it == nIter) break;
    }
  }

  // ---- epilogue: C layout VGPR r -> (M = base + r + 8*hi, N = l16) ----
  #pragma unroll
  for (int mt = 0; mt < 4; ++mt) {
    #pragma unroll
    for (int nt = 0; nt < 2; ++nt) {
      const int col = n0 + wn * 32 + nt * 16 + l16;
      if (col < N) {
        const float bv = (bias != nullptr && blockIdx.z == 0) ? bias[col] : 0.f;
        #pragma unroll
        for (int r = 0; r < 8; ++r) {
          const int row = m0 + wm * 64 + mt * 16 + hi * 8 + r;
          float* cp = &C[(size_t)row * (size_t)ldc + col];
          const float v = acc[mt][nt][r] + bv;
          if (split) unsafeAtomicAdd(cp, v);   // global_atomic_add_f32
          else       *cp = v;
        }
      }
    }
  }
}

// ---------------------------------------------------------------------------
__global__ __launch_bounds__(256)
void cvt_f32_bf16(const float* __restrict__ src, unsigned short* __restrict__ dst,
                  int n4)   // n4 = element count / 4 (count is multiple of 4)
{
  const int i = blockIdx.x * 256 + threadIdx.x;
  if (i < n4) {
    const float4 v = ((const float4*)src)[i];
    uint2 pk;
    pk.x = f2bf(v.x) | (f2bf(v.y) << 16);
    pk.y = f2bf(v.z) | (f2bf(v.w) << 16);
    ((uint2*)dst)[i] = pk;
  }
}

// ---------------------------------------------------------------------------
// Block reductions (LDS tree)
// ---------------------------------------------------------------------------
__device__ __forceinline__ float blk_max(float v, float* sh, int tid, int n) {
  sh[tid] = v; __syncthreads();
  for (int s = n >> 1; s > 0; s >>= 1) {
    if (tid < s) sh[tid] = fmaxf(sh[tid], sh[tid + s]);
    __syncthreads();
  }
  float r = sh[0]; __syncthreads(); return r;
}
__device__ __forceinline__ float blk_sum(float v, float* sh, int tid, int n) {
  sh[tid] = v; __syncthreads();
  for (int s = n >> 1; s > 0; s >>= 1) {
    if (tid < s) sh[tid] += sh[tid + s];
    __syncthreads();
  }
  float r = sh[0]; __syncthreads(); return r;
}
__device__ __forceinline__ int blk_argmax(float v, int idx, float* shf, int* shi,
                                          int tid, int n) {
  shf[tid] = v; shi[tid] = idx; __syncthreads();
  for (int s = n >> 1; s > 0; s >>= 1) {
    if (tid < s) {
      const float ov = shf[tid + s]; const int oi = shi[tid + s];
      if (ov > shf[tid] || (ov == shf[tid] && oi < shi[tid])) {
        shf[tid] = ov; shi[tid] = oi;
      }
    }
    __syncthreads();
  }
  int r = shi[0]; __syncthreads(); return r;
}

// ---------------------------------------------------------------------------
__global__ __launch_bounds__(256)
void rowdot(const float* __restrict__ Y, const float* __restrict__ ctx,
            float* __restrict__ pcpy)
{
  __shared__ float shf[256];
  const int t = blockIdx.x, tid = threadIdx.x;
  float p = 0.f;
  #pragma unroll
  for (int j = 0; j < 4; ++j) {
    const int idx = tid + j * 256;
    p += Y[(size_t)t * HS + idx] * ctx[(size_t)t * HS + idx];
  }
  const float s = blk_sum(p, shf, tid, 256);
  if (tid == 0) pcpy[t] = s;
}

// ---------------------------------------------------------------------------
// Special head row (2000 cols, register-resident). Writes sft as bf16 into the
// front half of its own fp32 row (no cross-row aliasing; read-all-then-write).
// ---------------------------------------------------------------------------
__global__ __launch_bounds__(256)
void rowproc_spt(float* __restrict__ buf, const float* __restrict__ g,
                 const int* __restrict__ tgt_ids, int* __restrict__ yspt)
{
  __shared__ float shf[256];
  __shared__ int   shi[256];
  const int t = blockIdx.x, tid = threadIdx.x;
  const float NEG = -__builtin_inff();
  float z[8];
  float lm = NEG;
  #pragma unroll
  for (int j = 0; j < 8; ++j) {
    const int idx = tid + j * 256;
    z[j] = (idx < NTGT) ? buf[(size_t)t * NTGT + idx] : NEG;
    lm = fmaxf(lm, z[j]);
  }
  const float m = blk_max(lm, shf, tid, 256);
  float ls = 0.f;
  #pragma unroll
  for (int j = 0; j < 8; ++j) {
    const int idx = tid + j * 256;
    if (idx < NTGT) ls += __expf(z[j] - m);
  }
  const float s = blk_sum(ls, shf, tid, 256);
  float lm2 = NEG;
  #pragma unroll
  for (int j = 0; j < 8; ++j) {
    const int idx = tid + j * 256;
    if (idx < NTGT) {
      z[j] = __logf(__expf(z[j] - m) / s + EPSV) + g[(size_t)t * NTGT + idx];
      lm2 = fmaxf(lm2, z[j]);
    } else z[j] = NEG;
  }
  const float m2 = blk_max(lm2, shf, tid, 256);
  float ls2 = 0.f;
  #pragma unroll
  for (int j = 0; j < 8; ++j) {
    const int idx = tid + j * 256;
    if (idx < NTGT) ls2 += __expf(z[j] - m2);
  }
  const float s2 = blk_sum(ls2, shf, tid, 256);
  unsigned short* obf = (unsigned short*)(buf + (size_t)t * NTGT);
  float bv = NEG; int bi = 0x7fffffff;
  #pragma unroll
  for (int j = 0; j < 8; ++j) {
    const int idx = tid + j * 256;
    if (idx < NTGT) {
      obf[idx] = (unsigned short)f2bf(__expf(z[j] - m2) / s2);
      if (z[j] > bv || (z[j] == bv && idx < bi)) { bv = z[j]; bi = idx; }
    }
  }
  const int gi = blk_argmax(bv, bi, shf, shi, tid, 256);
  if (tid == 0) yspt[t] = tgt_ids[gi];
}

// ---------------------------------------------------------------------------
// Normal head row (32000 vocab + copy score at index 32000), 1024 threads.
// ---------------------------------------------------------------------------
__global__ __launch_bounds__(1024)
void rowproc_nrm(float* __restrict__ buf, const float* __restrict__ g,
                 const float* __restrict__ pcpy, const int* __restrict__ inp_word,
                 float* __restrict__ sft_last, int* __restrict__ ynrm)
{
  __shared__ float shf[1024];
  __shared__ int   shi[1024];
  const int t = blockIdx.x, tid = threadIdx.x;
  const int V = NWRD + 1;
  const float NEG = -__builtin_inff();
  float z[32];
  float lm = NEG;
  #pragma unroll
  for (int j = 0; j < 32; ++j) {
    const int idx = tid + j * 1024;
    float v;
    if (idx < NWRD)       v = buf[(size_t)t * NWRD + idx];
    else if (idx == NWRD) v = pcpy[t];
    else                  v = NEG;
    z[j] = v; lm = fmaxf(lm, v);
  }
  const float m = blk_max(lm, shf, tid, 1024);
  float ls = 0.f;
  #pragma unroll
  for (int j = 0; j < 32; ++j) {
    const int idx = tid + j * 1024;
    if (idx < V) ls += __expf(z[j] - m);
  }
  const float s = blk_sum(ls, shf, tid, 1024);
  float lm2 = NEG;
  #pragma unroll
  for (int j = 0; j < 32; ++j) {
    const int idx = tid + j * 1024;
    if (idx < V) {
      z[j] = __logf(__expf(z[j] - m) / s + EPSV) + g[(size_t)t * V + idx];
      lm2 = fmaxf(lm2, z[j]);
    } else z[j] = NEG;
  }
  const float m2 = blk_max(lm2, shf, tid, 1024);
  float ls2 = 0.f;
  #pragma unroll
  for (int j = 0; j < 32; ++j) {
    const int idx = tid + j * 1024;
    if (idx < V) ls2 += __expf(z[j] - m2);
  }
  const float s2 = blk_sum(ls2, shf, tid, 1024);
  unsigned short* obf = (unsigned short*)(buf + (size_t)t * NWRD);
  float bv = NEG; int bi = 0x7fffffff;
  #pragma unroll
  for (int j = 0; j < 32; ++j) {
    const int idx = tid + j * 1024;
    if (idx < V) {
      const float sf = __expf(z[j] - m2) / s2;
      if (idx < NWRD)       obf[idx] = (unsigned short)f2bf(sf);
      else /* idx==NWRD */  sft_last[t] = sf;
      if (z[j] > bv || (z[j] == bv && idx < bi)) { bv = z[j]; bi = idx; }
    }
  }
  const int gi = blk_argmax(bv, bi, shf, shi, tid, 1024);
  if (tid == 0) ynrm[t] = (gi == NWRD) ? inp_word[t] : gi;
}

// ---------------------------------------------------------------------------
__global__ void zero_f32(float* __restrict__ p, int n) {
  const int i = blockIdx.x * blockDim.x + threadIdx.x;
  if (i < n) p[i] = 0.f;
}

// ---------------------------------------------------------------------------
__global__ __launch_bounds__(128)
void finalize(const int* __restrict__ inp_word, const float* __restrict__ masks,
              const int* __restrict__ spt_mask, const int* __restrict__ yspt,
              const int* __restrict__ ynrm, const float* __restrict__ sft_last,
              const float* __restrict__ sptE, const float* __restrict__ nrmE,
              const float* __restrict__ word_emb, float* __restrict__ out)
{
  const int t = blockIdx.x, d = threadIdx.x;
  const int iw = inp_word[t];
  const float sptv = (float)spt_mask[t];
  const float nrmv = (1.f - sptv) * masks[t];
  int word; float e;
  if (sptv > 0.f) {
    word = yspt[t];
    e = sptE[(size_t)t * DIM + d];
  } else if (nrmv > 0.f) {
    word = ynrm[t];
    e = word_emb[(size_t)iw * DIM + d] * sft_last[t] + nrmE[(size_t)t * DIM + d];
  } else {
    word = iw;
    e = word_emb[(size_t)iw * DIM + d];
  }
  out[(size_t)t * DIM + d] = e;
  if (d == 0) {
    out[(size_t)NT * DIM + t]      = (float)word;
    out[(size_t)NT * DIM + NT + t] = sptv;
  }
}

// ---------------------------------------------------------------------------
extern "C" void kernel_launch(void* const* d_in, const int* in_sizes, int n_in,
                              void* d_out, int out_size, void* d_ws, size_t ws_size,
                              hipStream_t stream)
{
  (void)in_sizes; (void)n_in; (void)out_size; (void)ws_size;
  const int*   inp_word     = (const int*)d_in[0];
  const float* masks        = (const float*)d_in[1];
  const int*   spt_mask     = (const int*)d_in[2];
  const int*   tgt_ids      = (const int*)d_in[3];
  const float* ctx          = (const float*)d_in[4];
  const float* W_spt        = (const float*)d_in[5];
  const float* b_spt        = (const float*)d_in[6];
  const float* W_nrm        = (const float*)d_in[7];
  const float* b_nrm        = (const float*)d_in[8];
  const float* W_cpy        = (const float*)d_in[9];
  const float* b_cpy        = (const float*)d_in[10];
  const float* word_emb     = (const float*)d_in[11];
  const float* word_emb_tgt = (const float*)d_in[12];
  const float* g_spt        = (const float*)d_in[13];
  const float* g_nrm        = (const float*)d_in[14];
  float* out = (float*)d_out;

  // workspace layout (floats): ~155 MB total
  float* ws    = (float*)d_ws;
  float* Ycpy  = ws;                          // NT*HS
  float* Lspt  = Ycpy + (size_t)NT * HS;      // NT*NTGT  (logits f32 -> sft bf16 in front half)
  float* Lnrm  = Lspt + (size_t)NT * NTGT;    // NT*NWRD  (logits f32 -> sft bf16 in front half)
  float* pcpy  = Lnrm + (size_t)NT * NWRD;    // NT
  float* sftl  = pcpy + NT;                   // NT
  int*   yspt  = (int*)(sftl + NT);           // NT
  int*   ynrm  = yspt + NT;                   // NT
  float* sptE  = (float*)(ynrm + NT);         // NT*DIM
  float* nrmE  = sptE + (size_t)NT * DIM;     // NT*DIM
  unsigned short* ctx_bf   = (unsigned short*)(nrmE + (size_t)NT * DIM);  // NT*HS
  unsigned short* wembT_bf = ctx_bf + (size_t)NT * HS;                    // NTGT*DIM
  unsigned short* wemb_bf  = wembT_bf + (size_t)NTGT * DIM;               // NWRD*DIM

  const dim3 blk(256);

  // bf16 pre-conversion of WMMA operands
  cvt_f32_bf16<<<(NT * HS / 4 + 255) / 256, blk, 0, stream>>>(ctx, ctx_bf, NT * HS / 4);
  cvt_f32_bf16<<<(NTGT * DIM / 4 + 255) / 256, blk, 0, stream>>>(
      word_emb_tgt, wembT_bf, NTGT * DIM / 4);
  cvt_f32_bf16<<<(NWRD * DIM / 4 + 255) / 256, blk, 0, stream>>>(
      word_emb, wemb_bf, NWRD * DIM / 4);

  // G1: Ycpy = ctx @ W_cpy + b_cpy
  gemm_wmma<false><<<dim3(HS / BN, NT / BM, 1), blk, 0, stream>>>(
      ctx_bf, W_cpy, b_cpy, Ycpy, NT, HS, HS, HS, HS, HS, HS);
  // G2: Lspt = ctx @ W_spt + b_spt
  gemm_wmma<false><<<dim3((NTGT + BN - 1) / BN, NT / BM, 1), blk, 0, stream>>>(
      ctx_bf, W_spt, b_spt, Lspt, NT, NTGT, HS, HS, NTGT, NTGT, HS);
  // G3: Lnrm = ctx @ W_nrm + b_nrm
  gemm_wmma<false><<<dim3(NWRD / BN, NT / BM, 1), blk, 0, stream>>>(
      ctx_bf, W_nrm, b_nrm, Lnrm, NT, NWRD, HS, HS, NWRD, NWRD, HS);

  rowdot<<<NT, 256, 0, stream>>>(Ycpy, ctx, pcpy);
  rowproc_spt<<<NT, 256, 0, stream>>>(Lspt, g_spt, tgt_ids, yspt);
  rowproc_nrm<<<NT, 1024, 0, stream>>>(Lnrm, g_nrm, pcpy, inp_word, sftl, ynrm);

  // zero split-K accumulators (sptE and nrmE are contiguous)
  zero_f32<<<(2 * NT * DIM + 255) / 256, blk, 0, stream>>>(sptE, 2 * NT * DIM);

  // G4: sptE += sft_spt_bf @ word_emb_tgt_bf   (split-K=8, Kc=256)
  gemm_wmma<true><<<dim3(1, NT / BM, 8), blk, 0, stream>>>(
      (const unsigned short*)Lspt, wembT_bf, nullptr, sptE,
      NT, DIM, NTGT, 2 * NTGT, DIM, DIM, 256);
  // G5: nrmE += sft_nrm_bf @ word_emb_bf       (split-K=16, Kc=2016)
  gemm_wmma<true><<<dim3(1, NT / BM, 16), blk, 0, stream>>>(
      (const unsigned short*)Lnrm, wemb_bf, nullptr, nrmE,
      NT, DIM, NWRD, 2 * NWRD, DIM, DIM, 2016);

  finalize<<<NT, DIM, 0, stream>>>(inp_word, masks, spt_mask, yspt, ynrm, sftl,
                                   sptE, nrmE, word_emb, out);
}
